// LocalizedPromptedAttentionLayer_46110768890017
// MI455X (gfx1250) — matrix operand order repaired and checked by reading, not compiled, not run on hardware
//
#include <hip/hip_runtime.h>
#include <hip/hip_bf16.h>
#include <math.h>

// ---------------------------------------------------------------------------
// CDNA5 (gfx1250) wave32 WMMA helpers: v_wmma_f32_16x16x32_bf16
// ---------------------------------------------------------------------------
typedef __attribute__((ext_vector_type(16))) __bf16 v16bf;
typedef __attribute__((ext_vector_type(8)))  __bf16 v8bf;
typedef __attribute__((ext_vector_type(8)))  float  v8f;

union AFrag { v16bf v; v8bf h[2]; };

// A-matrix fragment (16x32 bf16, M x K). ISA 7.12.2: lane L holds row M=L%16,
// K-chunks [kb..kb+7] and [kb+16..kb+23] with kb = (L<16 ? 0 : 8).
__device__ __forceinline__ v16bf load_frag_a(const __bf16* p, int stride, int k0) {
  int lane = threadIdx.x & 31;
  int row  = lane & 15;
  int kb   = k0 + ((lane & 16) ? 8 : 0);
  AFrag u;
  u.h[0] = *(const v8bf*)(p + (size_t)row * stride + kb);
  u.h[1] = *(const v8bf*)(p + (size_t)row * stride + kb + 16);
  return u.v;
}

// B-matrix fragment (32x16 bf16, K x N) from a [k][n] row-major buffer:
// lane L holds column N = n0+L%16, K-chunks [kb..kb+7] and [kb+16..kb+23], kb = (L<16 ? 0 : 8).
__device__ __forceinline__ v16bf load_frag_b(const __bf16* p, int stride, int k0, int n0) {
  int lane = threadIdx.x & 31;
  int n  = n0 + (lane & 15);
  int kb = k0 + ((lane & 16) ? 8 : 0);
  v16bf r;
  #pragma unroll
  for (int s = 0; s < 8; ++s) {
    r[s]     = p[(size_t)(kb + s) * stride + n];
    r[s + 8] = p[(size_t)(kb + 16 + s) * stride + n];
  }
  return r;
}

__device__ __forceinline__ v8f wmma_bf16(v16bf a, v16bf b, v8f c) {
  return __builtin_amdgcn_wmma_f32_16x16x32_bf16(false, a, false, b, (short)0, c, false, false);
}

// ---- gfx1250 async global->LDS copies (ASYNCcnt path, ISA §10 / §15.18.3) ---
// LDS aperture: flat LDS address truncates to the LDS byte offset (addr[31:0]).
__device__ __forceinline__ unsigned lds_off(const void* p) {
  return (unsigned)(unsigned long long)p;
}
__device__ __forceinline__ void async_copy_b128(unsigned lds_byte, const void* gptr) {
  asm volatile("global_load_async_to_lds_b128 %0, %1, off"
               :: "v"(lds_byte), "v"(gptr) : "memory");
}
__device__ __forceinline__ void async_copy_b64(unsigned lds_byte, const void* gptr) {
  asm volatile("global_load_async_to_lds_b64 %0, %1, off"
               :: "v"(lds_byte), "v"(gptr) : "memory");
}
__device__ __forceinline__ void wait_async0() {
  asm volatile("s_wait_asynccnt 0x0" ::: "memory");
}

// ---------------------------------------------------------------------------
// Problem constants
// ---------------------------------------------------------------------------
#define BB   1024
#define DD   512
#define NH   8
#define DH   64
#define LQX  26
#define LKY  256
#define NOUT 512

enum { EPI_L2N = 0, EPI_BIAS = 1, EPI_LN = 2 };

// GEMM dynamic-LDS layout: 2 x 32KB B slabs + 2 x 2KB A tiles
#define LB_BYTES   (32 * NOUT * 2)                  // 32 KB
#define LA_BYTES   (32 * 32 * 2)                    // 2 KB
#define GEMM_LDS   (2 * LB_BYTES + 2 * LA_BYTES)    // 69632 B (CDNA5: 320KB/WGP)

// ---------------------------------------------------------------------------
// Weight prep: WT[k][n] = (bf16) W[n][k]   (W is [NOUT x Kdim] row-major)
// ---------------------------------------------------------------------------
__global__ __launch_bounds__(256)
void prep_weights(const float* __restrict__ W, __bf16* __restrict__ WT, int Kdim) {
  int idx = blockIdx.x * 256 + threadIdx.x;
  if (idx < Kdim * NOUT) {
    int k = idx / NOUT, n = idx - k * NOUT;
    WT[idx] = (__bf16)W[(size_t)n * Kdim + k];
  }
}

// ---------------------------------------------------------------------------
// Row-block WMMA GEMM:  C[32 x 512] = A[32 x K] * WT[K x 512]  (+ epilogue)
// Double-buffered async pipeline:
//   iter ks: s_wait_asynccnt 0 (slab ks landed) -> barrier ->
//            issue async for ks+1 into other buffer -> WMMA on slab ks
// ---------------------------------------------------------------------------
template <bool A_IS_BF16, int EPI>
__global__ __launch_bounds__(256)
void gemm_rowblock(const void* __restrict__ Aptr,
                   const __bf16* __restrict__ WT,   // [Kdim][512] bf16
                   const float* __restrict__ bias,
                   int Kdim, int nrows,
                   __bf16* __restrict__ outBf,
                   float* __restrict__ outF32,
                   const float* __restrict__ resid,
                   const float* __restrict__ gamma,
                   const float* __restrict__ beta,
                   float* __restrict__ outLN) {
  extern __shared__ char smem[];
  // NOTE: compute buffer pointers on demand (no constant pointer aggregates
  // from the extern-LDS symbol -> avoids ld.lld static-initializer error).
  auto lB = [&](int buf) -> __bf16* {
    return (__bf16*)(smem + (size_t)buf * LB_BYTES);
  };
  auto lA = [&](int buf) -> __bf16* {
    return (__bf16*)(smem + 2 * LB_BYTES + (size_t)buf * LA_BYTES);
  };
  __shared__ float red[32];
  __shared__ float red2[32];

  const int row0 = blockIdx.x * 32;
  const int wave = threadIdx.x >> 5;
  const int lane = threadIdx.x & 31;
  const int halfm = (lane >> 4) * 8;
  const int nloc  = lane & 15;
  const int nsteps = Kdim >> 5;

  v8f acc[2][4];
  #pragma unroll
  for (int mt = 0; mt < 2; ++mt)
    #pragma unroll
    for (int nt = 0; nt < 4; ++nt) acc[mt][nt] = (v8f)0.0f;

  // ---- staging helpers ----
  auto issueB = [&](int ks, int buf) {
    unsigned base = lds_off(lB(buf));
    const char* g = (const char*)WT + (size_t)ks * 32 * NOUT * 2;
    #pragma unroll
    for (int j = 0; j < 8; ++j) {
      unsigned u = threadIdx.x + j * 256;
      async_copy_b128(base + u * 16, g + (size_t)u * 16);
    }
  };
  auto issueA_bf16 = [&](int ks, int buf) {
    unsigned u = threadIdx.x;
    int r = u >> 3, c = u & 7;              // 8 x 8B chunks per 32-elem row
    async_copy_b64(lds_off(lA(buf)) + r * 64 + c * 8,
                   (const char*)Aptr + 2 * ((size_t)(row0 + r) * Kdim + ks * 32) + c * 8);
  };
  auto stageA_f32 = [&](int ks, int buf) {
    __bf16* dst = lA(buf);
    for (int i = threadIdx.x; i < 32 * 32; i += 256) {
      int m = i >> 5, k = i & 31;
      int row = row0 + m;
      float v = (row < nrows) ? ((const float*)Aptr)[(size_t)row * Kdim + ks * 32 + k]
                              : 0.0f;
      dst[i] = (__bf16)v;
    }
  };

  // ---- prologue: stage step 0 into buffer 0 ----
  issueB(0, 0);
  if (A_IS_BF16) issueA_bf16(0, 0); else stageA_f32(0, 0);

  // ---- pipelined main loop ----
  for (int ks = 0; ks < nsteps; ++ks) {
    const int cur = ks & 1;
    wait_async0();                 // slab ks landed (in-order per wave)
    __syncthreads();               // cross-wave visibility; other buffer free
    if (ks + 1 < nsteps) {
      issueB(ks + 1, 1 - cur);     // DMA next slab while we compute
      if (A_IS_BF16) issueA_bf16(ks + 1, 1 - cur);
      else           stageA_f32(ks + 1, 1 - cur);
    }
    const __bf16* lAc = lA(cur);
    const __bf16* lBc = lB(cur);
    #pragma unroll
    for (int mt = 0; mt < 2; ++mt) {
      v16bf af = load_frag_a(lAc + mt * 16 * 32, 32, 0);
      #pragma unroll
      for (int nt = 0; nt < 4; ++nt) {
        v16bf bf = load_frag_b(lBc, NOUT, 0, wave * 64 + nt * 16);
        acc[mt][nt] = wmma_bf16(af, bf, acc[mt][nt]);
      }
    }
  }

  // ---------------- epilogues ----------------
  #pragma unroll
  for (int mt = 0; mt < 2; ++mt)
    #pragma unroll
    for (int nt = 0; nt < 4; ++nt) {
      float bv = bias ? bias[wave * 64 + nt * 16 + nloc] : 0.0f;
      #pragma unroll
      for (int i = 0; i < 8; ++i) acc[mt][nt][i] += bv;
    }

  if (EPI == EPI_L2N) {
    __syncthreads();
    if (threadIdx.x < 32) red[threadIdx.x] = 0.0f;
    __syncthreads();
    #pragma unroll
    for (int mt = 0; mt < 2; ++mt)
      #pragma unroll
      for (int nt = 0; nt < 4; ++nt)
        #pragma unroll
        for (int i = 0; i < 8; ++i)
          atomicAdd(&red[mt * 16 + halfm + i], acc[mt][nt][i] * acc[mt][nt][i]);
    __syncthreads();
    #pragma unroll
    for (int mt = 0; mt < 2; ++mt)
      #pragma unroll
      for (int nt = 0; nt < 4; ++nt) {
        int col = wave * 64 + nt * 16 + nloc;
        #pragma unroll
        for (int i = 0; i < 8; ++i) {
          int m = mt * 16 + halfm + i, row = row0 + m;
          if (row < nrows) {
            float s = 1.0f / (sqrtf(red[m]) + 1e-6f);
            float v = acc[mt][nt][i] * s;
            outBf[(size_t)row * NOUT + col] = (__bf16)v;
            if (outF32) outF32[(size_t)row * NOUT + col] = v;
          }
        }
      }
  } else if (EPI == EPI_BIAS) {
    #pragma unroll
    for (int mt = 0; mt < 2; ++mt)
      #pragma unroll
      for (int nt = 0; nt < 4; ++nt) {
        int col = wave * 64 + nt * 16 + nloc;
        #pragma unroll
        for (int i = 0; i < 8; ++i) {
          int m = mt * 16 + halfm + i, row = row0 + m;
          if (row < nrows)
            outBf[(size_t)row * NOUT + col] = (__bf16)acc[mt][nt][i];
        }
      }
  } else { // EPI_LN
    __syncthreads();
    if (threadIdx.x < 32) { red[threadIdx.x] = 0.0f; red2[threadIdx.x] = 0.0f; }
    __syncthreads();
    #pragma unroll
    for (int mt = 0; mt < 2; ++mt)
      #pragma unroll
      for (int nt = 0; nt < 4; ++nt) {
        int col = wave * 64 + nt * 16 + nloc;
        #pragma unroll
        for (int i = 0; i < 8; ++i) {
          int m = mt * 16 + halfm + i, row = row0 + m;
          int b = row / LQX, r = row % LQX;
          int xr = b * LQX + (r < 16 ? 10 + r : r - 16);
          float v = acc[mt][nt][i] + resid[(size_t)xr * NOUT + col];
          acc[mt][nt][i] = v;
          atomicAdd(&red[m], v);
          atomicAdd(&red2[m], v * v);
        }
      }
    __syncthreads();
    #pragma unroll
    for (int mt = 0; mt < 2; ++mt)
      #pragma unroll
      for (int nt = 0; nt < 4; ++nt) {
        int col = wave * 64 + nt * 16 + nloc;
        #pragma unroll
        for (int i = 0; i < 8; ++i) {
          int m = mt * 16 + halfm + i, row = row0 + m;
          if (row < nrows) {
            float mu  = red[m] * (1.0f / NOUT);
            float var = red2[m] * (1.0f / NOUT) - mu * mu;
            float inv = rsqrtf(var + 1e-6f);
            outLN[(size_t)row * NOUT + col] =
                gamma[col] * (acc[mt][nt][i] - mu) * inv + beta[col];
          }
        }
      }
  }
}

// ---------------------------------------------------------------------------
// Local windowed attention: one wave32 per window (B*16 windows)
// ---------------------------------------------------------------------------
__global__ __launch_bounds__(32)
void local_attn_kernel(const __bf16* __restrict__ Q,
                       const __bf16* __restrict__ K,
                       const __bf16* __restrict__ V,
                       __bf16* __restrict__ ctx) {
  int win = blockIdx.x;
  int b = win >> 4, r = win & 15;
  int lane = threadIdx.x;
  size_t qrow  = (size_t)(b * LQX + 10 + r) * DD;
  size_t kbase = (size_t)(b * LKY + r * 16) * DD;
  size_t crow  = (size_t)(b * LQX + r) * DD;

  for (int h = 0; h < NH; ++h) {
    int off = h * DH;
    int j = lane & 15;
    float s = 0.0f;
    #pragma unroll 8
    for (int d = 0; d < DH; ++d)
      s += (float)Q[qrow + off + d] * (float)K[kbase + (size_t)j * DD + off + d];
    s *= 0.125f;
    float mx = s;
    for (int x = 8; x; x >>= 1) mx = fmaxf(mx, __shfl_xor(mx, x, 16));
    float p = __expf(s - mx);
    float sum = p;
    for (int x = 8; x; x >>= 1) sum += __shfl_xor(sum, x, 16);
    p /= sum;

    int d0 = lane * 2;
    float o0 = 0.0f, o1 = 0.0f;
    #pragma unroll
    for (int jj = 0; jj < 16; ++jj) {
      float pj = __shfl(p, jj, 16);
      o0 += pj * (float)V[kbase + (size_t)jj * DD + off + d0];
      o1 += pj * (float)V[kbase + (size_t)jj * DD + off + d0 + 1];
    }
    ctx[crow + off + d0]     = (__bf16)o0;
    ctx[crow + off + d0 + 1] = (__bf16)o1;
  }
}

// ---------------------------------------------------------------------------
// Trainable attention: one block per (batch, head); 4 waves; WMMA both GEMMs.
// ---------------------------------------------------------------------------
__global__ __launch_bounds__(128)
void trainable_attn_kernel(const __bf16* __restrict__ Q,
                           const __bf16* __restrict__ K,
                           const __bf16* __restrict__ V,
                           __bf16* __restrict__ ctx) {
  __shared__ __bf16 lQ[16 * DH];
  __shared__ __bf16 lKT[DH * LKY];
  __shared__ __bf16 lP[16 * LKY];
  __shared__ float  rmax[4][16], rsum[4][16];
  __shared__ float  rowmax[16], rowsum[16];

  int b = blockIdx.x >> 3, h = blockIdx.x & 7;
  int wave = threadIdx.x >> 5, lane = threadIdx.x & 31;
  int halfm = (lane >> 4) * 8, nloc = lane & 15;

  for (int i = threadIdx.x; i < 16 * DH; i += 128) {
    int m = i >> 6, d = i & 63;
    lQ[i] = (m < 10) ? Q[(size_t)(b * LQX + m) * DD + h * DH + d] : (__bf16)0.0f;
  }
  for (int i = threadIdx.x; i < DH * LKY; i += 128) {
    int key = i >> 6, d = i & 63;
    lKT[d * LKY + key] = K[(size_t)(b * LKY + key) * DD + h * DH + d];
  }
  __syncthreads();

  v8f sc[4];
  #pragma unroll
  for (int nt = 0; nt < 4; ++nt) sc[nt] = (v8f)0.0f;
  #pragma unroll
  for (int k0 = 0; k0 < DH; k0 += 32) {
    v16bf af = load_frag_a(lQ, DH, k0);
    #pragma unroll
    for (int nt = 0; nt < 4; ++nt)
      sc[nt] = wmma_bf16(af, load_frag_b(lKT, LKY, k0, (wave * 4 + nt) * 16), sc[nt]);
  }
  #pragma unroll
  for (int nt = 0; nt < 4; ++nt)
    #pragma unroll
    for (int i = 0; i < 8; ++i) sc[nt][i] *= 0.125f;

  float lm[8];
  #pragma unroll
  for (int i = 0; i < 8; ++i) {
    lm[i] = fmaxf(fmaxf(sc[0][i], sc[1][i]), fmaxf(sc[2][i], sc[3][i]));
    for (int x = 1; x < 16; x <<= 1) lm[i] = fmaxf(lm[i], __shfl_xor(lm[i], x));
  }
  if (nloc == 0)
    #pragma unroll
    for (int i = 0; i < 8; ++i) rmax[wave][halfm + i] = lm[i];
  __syncthreads();
  if (threadIdx.x < 16) {
    float v = -1e30f;
    for (int w = 0; w < 4; ++w) v = fmaxf(v, rmax[w][threadIdx.x]);
    rowmax[threadIdx.x] = v;
  }
  __syncthreads();

  float ls[8];
  #pragma unroll
  for (int i = 0; i < 8; ++i) ls[i] = 0.0f;
  #pragma unroll
  for (int nt = 0; nt < 4; ++nt)
    #pragma unroll
    for (int i = 0; i < 8; ++i) {
      float p = __expf(sc[nt][i] - rowmax[halfm + i]);
      sc[nt][i] = p;
      ls[i] += p;
    }
  #pragma unroll
  for (int i = 0; i < 8; ++i)
    for (int x = 1; x < 16; x <<= 1) ls[i] += __shfl_xor(ls[i], x);
  if (nloc == 0)
    #pragma unroll
    for (int i = 0; i < 8; ++i) rsum[wave][halfm + i] = ls[i];
  __syncthreads();
  if (threadIdx.x < 16) {
    float v = 0.0f;
    for (int w = 0; w < 4; ++w) v += rsum[w][threadIdx.x];
    rowsum[threadIdx.x] = v;
  }
  __syncthreads();

  #pragma unroll
  for (int nt = 0; nt < 4; ++nt) {
    int n = (wave * 4 + nt) * 16 + nloc;
    #pragma unroll
    for (int i = 0; i < 8; ++i) {
      int m = halfm + i;
      lP[m * LKY + n] = (__bf16)(sc[nt][i] / rowsum[m]);
    }
  }
  __syncthreads();

  const __bf16* Vbase = V + (size_t)(b * LKY) * DD + h * DH;
  v8f o = (v8f)0.0f;
  #pragma unroll
  for (int k0 = 0; k0 < LKY; k0 += 32) {
    v16bf af = load_frag_a(lP, LKY, k0);
    v16bf bf = load_frag_b(Vbase, DD, k0, wave * 16);
    o = wmma_bf16(af, bf, o);
  }
  #pragma unroll
  for (int i = 0; i < 8; ++i) {
    int m = halfm + i;
    if (m < 10)
      ctx[(size_t)(b * LQX + 16 + m) * DD + h * DH + wave * 16 + nloc] = (__bf16)o[i];
  }
}

// ---------------------------------------------------------------------------
// Host-side launch
// ---------------------------------------------------------------------------
extern "C" void kernel_launch(void* const* d_in, const int* in_sizes, int n_in,
                              void* d_out, int out_size, void* d_ws, size_t ws_size,
                              hipStream_t stream) {
  const float* x     = (const float*)d_in[0];
  const float* y     = (const float*)d_in[1];
  const float* Wx    = (const float*)d_in[2];
  const float* bx    = (const float*)d_in[3];
  const float* Wy    = (const float*)d_in[4];
  const float* by    = (const float*)d_in[5];
  const float* ipw   = (const float*)d_in[6];
  const float* ipb   = (const float*)d_in[7];
  const float* outw  = (const float*)d_in[8];
  const float* outb  = (const float*)d_in[9];
  const float* gamma = (const float*)d_in[10];
  const float* beta  = (const float*)d_in[11];
  float* out = (float*)d_out;

  const size_t nYrows = (size_t)BB * LKY;
  const size_t nXrows = (size_t)BB * LQX;

  size_t off = 0;
  auto carve = [&](size_t bytes) {
    void* p = (char*)d_ws + off;
    off += (bytes + 255) & ~(size_t)255;
    return p;
  };
  __bf16* yprojB = (__bf16*)carve(nYrows * DD * sizeof(__bf16));
  __bf16* Kb     = (__bf16*)carve(nYrows * DD * sizeof(__bf16));
  __bf16* Vb     = (__bf16*)carve(nYrows * DD * sizeof(__bf16));
  float*  xprojF = (float*) carve(nXrows * DD * sizeof(float));
  __bf16* xprojB = (__bf16*)carve(nXrows * DD * sizeof(__bf16));
  __bf16* Qb     = (__bf16*)carve(nXrows * DD * sizeof(__bf16));
  __bf16* ctx    = (__bf16*)carve(nXrows * DD * sizeof(__bf16));
  __bf16* WyT    = (__bf16*)carve((size_t)1024 * NOUT * sizeof(__bf16));
  __bf16* WxT    = (__bf16*)carve((size_t)1536 * NOUT * sizeof(__bf16));
  __bf16* WqT    = (__bf16*)carve((size_t)DD * NOUT * sizeof(__bf16));
  __bf16* WkT    = (__bf16*)carve((size_t)DD * NOUT * sizeof(__bf16));
  __bf16* WvT    = (__bf16*)carve((size_t)DD * NOUT * sizeof(__bf16));
  __bf16* WoT    = (__bf16*)carve((size_t)DD * NOUT * sizeof(__bf16));
  (void)ws_size; (void)in_sizes; (void)n_in; (void)out_size;

  prep_weights<<<dim3((1024 * NOUT + 255) / 256), dim3(256), 0, stream>>>(Wy, WyT, 1024);
  prep_weights<<<dim3((1536 * NOUT + 255) / 256), dim3(256), 0, stream>>>(Wx, WxT, 1536);
  prep_weights<<<dim3((DD * NOUT + 255) / 256), dim3(256), 0, stream>>>(ipw, WqT, DD);
  prep_weights<<<dim3((DD * NOUT + 255) / 256), dim3(256), 0, stream>>>(ipw + (size_t)DD * DD, WkT, DD);
  prep_weights<<<dim3((DD * NOUT + 255) / 256), dim3(256), 0, stream>>>(ipw + (size_t)2 * DD * DD, WvT, DD);
  prep_weights<<<dim3((DD * NOUT + 255) / 256), dim3(256), 0, stream>>>(outw, WoT, DD);

  gemm_rowblock<false, EPI_L2N><<<dim3(nYrows / 32), dim3(256), GEMM_LDS, stream>>>(
      y, WyT, by, 1024, (int)nYrows, yprojB, nullptr, nullptr, nullptr, nullptr, nullptr);
  gemm_rowblock<false, EPI_L2N><<<dim3(nXrows / 32), dim3(256), GEMM_LDS, stream>>>(
      x, WxT, bx, 1536, (int)nXrows, xprojB, xprojF, nullptr, nullptr, nullptr, nullptr);
  gemm_rowblock<true, EPI_BIAS><<<dim3(nXrows / 32), dim3(256), GEMM_LDS, stream>>>(
      xprojB, WqT, ipb, DD, (int)nXrows, Qb, nullptr, nullptr, nullptr, nullptr, nullptr);
  gemm_rowblock<true, EPI_BIAS><<<dim3(nYrows / 32), dim3(256), GEMM_LDS, stream>>>(
      yprojB, WkT, ipb + DD, DD, (int)nYrows, Kb, nullptr, nullptr, nullptr, nullptr, nullptr);
  gemm_rowblock<true, EPI_BIAS><<<dim3(nYrows / 32), dim3(256), GEMM_LDS, stream>>>(
      yprojB, WvT, ipb + 2 * DD, DD, (int)nYrows, Vb, nullptr, nullptr, nullptr, nullptr, nullptr);
  local_attn_kernel<<<dim3(BB * 16), dim3(32), 0, stream>>>(Qb, Kb, Vb, ctx);
  trainable_attn_kernel<<<dim3(BB * NH), dim3(128), 0, stream>>>(Qb, Kb, Vb, ctx);
  gemm_rowblock<true, EPI_LN><<<dim3(nXrows / 32), dim3(256), GEMM_LDS, stream>>>(
      ctx, WoT, outb, DD, (int)nXrows, nullptr, nullptr, xprojF, gamma, beta, out);
}
